// SelfAttention_31129922961544
// MI455X (gfx1250) — matrix operand order repaired and checked by reading, not compiled, not run on hardware
//
#include <hip/hip_runtime.h>
#include <hip/hip_bf16.h>

// ---------------------------------------------------------------------------
// Types for CDNA5 WMMA (wave32): v_wmma_f32_16x16x32_bf16
// ---------------------------------------------------------------------------
typedef __attribute__((ext_vector_type(16))) __bf16 v16bf;
typedef __attribute__((ext_vector_type(8)))  __bf16 v8bf;
typedef __attribute__((ext_vector_type(4)))  __bf16 v4bf;
typedef __attribute__((ext_vector_type(8)))  float  v8f;
typedef __attribute__((ext_vector_type(4)))  float  v4f;

#define D_MODEL 1024
#define HEADS   16
#define D_K     64
#define SEQ     2048
#define BATCH   4
#define ROWS    (BATCH * SEQ)          // 8192

__device__ __forceinline__ v8f wmma_bf16(v16bf a, v16bf b, v8f c) {
    return __builtin_amdgcn_wmma_f32_16x16x32_bf16(
        /*neg_a=*/false, a, /*neg_b=*/false, b,
        /*c_mod=*/(short)0, c, /*reuse_a=*/false, /*reuse_b=*/false);
}

union BfFrag { v16bf v; v8bf h[2]; };

// ---------------------------------------------------------------------------
// GEMM: C[M,N] = A(f32, M x K, row-major) * B(f32, K x N, row-major) + bias
// MODE 0: QKV projection -> scatter bf16 into q/k/v head-major (B,H,S,Dk)
// MODE 1: out projection -> res = acc + bias + residual(x), f32
// Workgroup: 256 thr (8 waves), tile 128(M) x 128(N), K step 32.
// Double-buffered LDS + register prefetch: global loads for step t+1 are
// issued before step t's WMMAs so HBM latency hides behind the matrix pipe.
// ---------------------------------------------------------------------------
#define AS_STR 40   // bf16 units; 80 B row stride (16B aligned)
#define BS_STR 40

template<int MODE>
__global__ __launch_bounds__(256)
void gemm_bf16_wmma(const float* __restrict__ A, const float* __restrict__ B,
                    const float* __restrict__ bias, const float* __restrict__ resid,
                    __bf16* __restrict__ qo, __bf16* __restrict__ ko,
                    __bf16* __restrict__ vo, float* __restrict__ res_out,
                    int M, int N, int K)
{
    __shared__ __bf16 As[2][128][AS_STR];   // [buf][m][k]
    __shared__ __bf16 Bs[2][128][BS_STR];   // [buf][n][k] (transposed)

    const int tid  = threadIdx.x;
    const int lane = tid & 31;
    const int wave = tid >> 5;
    const int half = lane >> 4;
    const int l16  = lane & 15;
    const int m0   = blockIdx.y * 128;
    const int n0   = blockIdx.x * 128;
    const int wm   = (wave >> 2) * 64;   // 0 / 64
    const int wn   = (wave & 3) * 32;    // 0..96

    v8f acc[4][2];
    const v8f vzero = {0.f,0.f,0.f,0.f,0.f,0.f,0.f,0.f};
#pragma unroll
    for (int mt = 0; mt < 4; ++mt)
#pragma unroll
        for (int nt = 0; nt < 2; ++nt) acc[mt][nt] = vzero;

    v4f fa[4], fb[4];

    // ---- prologue: fetch k-step 0 and stage into buffer 0 --------------
#pragma unroll
    for (int i = 0; i < 4; ++i) {
        int e = tid + i * 256;
        fa[i] = *(const v4f*)(A + (size_t)(m0 + (e >> 3)) * K + (e & 7) * 4);
        fb[i] = *(const v4f*)(B + (size_t)(e >> 5) * N + n0 + (e & 31) * 4);
    }
#pragma unroll
    for (int i = 0; i < 4; ++i) {
        int e = tid + i * 256;
        int row = e >> 3, qd = e & 7;
        v4bf b4;
        b4[0] = (__bf16)fa[i].x; b4[1] = (__bf16)fa[i].y;
        b4[2] = (__bf16)fa[i].z; b4[3] = (__bf16)fa[i].w;
        *(v4bf*)&As[0][row][qd * 4] = b4;
        int kr = e >> 5, nq = e & 31;
        Bs[0][nq * 4 + 0][kr] = (__bf16)fb[i].x;
        Bs[0][nq * 4 + 1][kr] = (__bf16)fb[i].y;
        Bs[0][nq * 4 + 2][kr] = (__bf16)fb[i].z;
        Bs[0][nq * 4 + 3][kr] = (__bf16)fb[i].w;
    }
    __syncthreads();

    const int NSTEP = K >> 5;
    int buf = 0;
    for (int step = 0; step < NSTEP; ++step) {
        const bool hasNext = (step + 1) < NSTEP;
        const int knext = (step + 1) << 5;

        // ---- prefetch next K-step's f32 quads into registers -----------
        if (hasNext) {
#pragma unroll
            for (int i = 0; i < 4; ++i) {
                int e = tid + i * 256;
                fa[i] = *(const v4f*)(A + (size_t)(m0 + (e >> 3)) * K + knext + (e & 7) * 4);
                fb[i] = *(const v4f*)(B + (size_t)(knext + (e >> 5)) * N + n0 + (e & 31) * 4);
            }
        }

        // ---- compute current buffer ------------------------------------
        // B fragments (32x16): K = 8h+j | 16+8h+j
        v16bf bfr[2];
#pragma unroll
        for (int nt = 0; nt < 2; ++nt) {
            int n = wn + nt * 16 + l16;
            BfFrag u;
            u.h[0] = *(const v8bf*)&Bs[buf][n][8 * half];
            u.h[1] = *(const v8bf*)&Bs[buf][n][16 + 8 * half];
            bfr[nt] = u.v;
        }
        // A fragments (16x32): K = 8h+j | 16+8h+j
#pragma unroll
        for (int mt = 0; mt < 4; ++mt) {
            int m = wm + mt * 16 + l16;
            BfFrag u;
            u.h[0] = *(const v8bf*)&As[buf][m][8 * half];
            u.h[1] = *(const v8bf*)&As[buf][m][16 + 8 * half];
#pragma unroll
            for (int nt = 0; nt < 2; ++nt)
                acc[mt][nt] = wmma_bf16(u.v, bfr[nt], acc[mt][nt]);
        }

        // ---- convert + store prefetched data into the other buffer -----
        if (hasNext) {
            int nb = buf ^ 1;
#pragma unroll
            for (int i = 0; i < 4; ++i) {
                int e = tid + i * 256;
                int row = e >> 3, qd = e & 7;
                v4bf b4;
                b4[0] = (__bf16)fa[i].x; b4[1] = (__bf16)fa[i].y;
                b4[2] = (__bf16)fa[i].z; b4[3] = (__bf16)fa[i].w;
                *(v4bf*)&As[nb][row][qd * 4] = b4;
                int kr = e >> 5, nq = e & 31;
                Bs[nb][nq * 4 + 0][kr] = (__bf16)fb[i].x;
                Bs[nb][nq * 4 + 1][kr] = (__bf16)fb[i].y;
                Bs[nb][nq * 4 + 2][kr] = (__bf16)fb[i].z;
                Bs[nb][nq * 4 + 3][kr] = (__bf16)fb[i].w;
            }
            __syncthreads();
        }
        buf ^= 1;
    }

    // ---- epilogue: C layout = lane column l16, VGPR i row i+8*half -----
#pragma unroll
    for (int mt = 0; mt < 4; ++mt)
#pragma unroll
        for (int nt = 0; nt < 2; ++nt)
#pragma unroll
            for (int i = 0; i < 8; ++i) {
                int m = m0 + wm + mt * 16 + 8 * half + i;
                int n = n0 + wn + nt * 16 + l16;
                float val = acc[mt][nt][i] + bias[n];
                if (MODE == 0) {
                    int which = n >> 10;            // 0=q 1=k 2=v
                    int d0 = n & 1023;
                    int h = d0 >> 6, d = d0 & 63;
                    int b = m >> 11, s = m & (SEQ - 1);
                    size_t off = (((size_t)(b * HEADS + h) * SEQ) + s) * D_K + d;
                    __bf16* dst = (which == 0) ? qo : (which == 1) ? ko : vo;
                    dst[off] = (__bf16)val;
                } else {
                    size_t off = (size_t)m * N + n;
                    res_out[off] = val + resid[off];
                }
            }
}

// ---------------------------------------------------------------------------
// Flash attention: grid = B*H*(S/128), 8 waves x 16 q-rows, 64-key chunks.
// q,k,v bf16 in (B,H,S,Dk); ctx f32 out in (B,S,D_MODEL).
// K tile staged via GLOBAL_LOAD_ASYNC_TO_LDS_B128 (ASYNCcnt path); V tile
// staged transposed by hand (async copies cannot transpose).
// ---------------------------------------------------------------------------
#define KV_STR 72   // 144 B row stride (16B aligned)

__global__ __launch_bounds__(256)
void flash_attn_wmma(const __bf16* __restrict__ q, const __bf16* __restrict__ k,
                     const __bf16* __restrict__ v, float* __restrict__ ctx)
{
    __shared__ __bf16 Ks[64][KV_STR];        // [key][d]
    __shared__ __bf16 Vs[64][KV_STR];        // [d][key]  (transposed)
    __shared__ __bf16 Ps[8][16][KV_STR];     // per-wave P patch [q][key]

    const int tid  = threadIdx.x;
    const int lane = tid & 31;
    const int wave = tid >> 5;
    const int half = lane >> 4;
    const int l16  = lane & 15;
    const int bh   = blockIdx.x >> 4;              // (b*H + h)
    const int q0   = (blockIdx.x & 15) * 128;
    const int b    = bh >> 4;
    const int h    = bh & (HEADS - 1);
    const size_t base = (size_t)bh * SEQ * D_K;
    const int wq   = wave * 16;
    const float softscale = 0.125f;                // 1/sqrt(64)

    // per-thread staging slot (fixed): 512 octets of 8 bf16 over 2 iters
    // Q fragments for this wave's 16 rows (two 16x32 frags over d)
    v16bf qf[2];
#pragma unroll
    for (int c = 0; c < 2; ++c) {
        const __bf16* qrow = q + base + (size_t)(q0 + wq + l16) * D_K;
        BfFrag u;
        u.h[0] = *(const v8bf*)(qrow + c * 32 + 8 * half);
        u.h[1] = *(const v8bf*)(qrow + c * 32 + 16 + 8 * half);
        qf[c] = u.v;
    }

    float mi[8], li[8];
    v8f accc[4];
    const v8f vzero = {0.f,0.f,0.f,0.f,0.f,0.f,0.f,0.f};
#pragma unroll
    for (int i = 0; i < 8; ++i) { mi[i] = -1e30f; li[i] = 0.f; }
#pragma unroll
    for (int dt = 0; dt < 4; ++dt) accc[dt] = vzero;

    for (int kt = 0; kt < SEQ; kt += 64) {
        // ---- K tile: async global->LDS copy (no VGPR round-trip) -------
#pragma unroll
        for (int i = 0; i < 2; ++i) {
            int e = tid + i * 256;                 // 512 octets of 8 bf16
            int key = e >> 3, d = (e & 7) * 8;
            unsigned ldsa = (unsigned)(size_t)&Ks[key][d];
            const __bf16* gptr = k + base + (size_t)(kt + key) * D_K + d;
            asm volatile("global_load_async_to_lds_b128 %0, %1, off"
                         :: "v"(ldsa), "v"(gptr) : "memory");
        }
        // ---- V tile: load + transpose into LDS -------------------------
#pragma unroll
        for (int i = 0; i < 2; ++i) {
            int e = tid + i * 256;
            int key = e >> 3, d = (e & 7) * 8;
            v8bf vv = *(const v8bf*)(v + base + (size_t)(kt + key) * D_K + d);
#pragma unroll
            for (int j = 0; j < 8; ++j) Vs[d + j][key] = vv[j];
        }
        asm volatile("s_wait_asynccnt 0x0" ::: "memory");
        __syncthreads();

        // ---- scores S = Q * K^T (4 tiles of 16 keys) -------------------
        v8f sc[4];
#pragma unroll
        for (int nt = 0; nt < 4; ++nt) {
            sc[nt] = vzero;
            int key = nt * 16 + l16;
#pragma unroll
            for (int c = 0; c < 2; ++c) {          // two 32-wide d chunks
                BfFrag u;
                u.h[0] = *(const v8bf*)&Ks[key][c * 32 + 8 * half];
                u.h[1] = *(const v8bf*)&Ks[key][c * 32 + 16 + 8 * half];
                sc[nt] = wmma_bf16(qf[c], u.v, sc[nt]);
            }
        }

        // ---- online softmax (row = i + 8*half, 16 cols in half-wave) ---
#pragma unroll
        for (int i = 0; i < 8; ++i) {
            float mx = fmaxf(fmaxf(sc[0][i], sc[1][i]), fmaxf(sc[2][i], sc[3][i]));
#pragma unroll
            for (int msk = 8; msk >= 1; msk >>= 1)
                mx = fmaxf(mx, __shfl_xor(mx, msk, 32));
            mx *= softscale;
            float mnew = fmaxf(mi[i], mx);
            float f = __expf(mi[i] - mnew);
            mi[i] = mnew;
            float rs = 0.f;
#pragma unroll
            for (int nt = 0; nt < 4; ++nt) {
                float p = __expf(sc[nt][i] * softscale - mnew);
                rs += p;
                Ps[wave][8 * half + i][nt * 16 + l16] = (__bf16)p;
            }
#pragma unroll
            for (int msk = 8; msk >= 1; msk >>= 1)
                rs += __shfl_xor(rs, msk, 32);
            li[i] = li[i] * f + rs;
#pragma unroll
            for (int dt = 0; dt < 4; ++dt) accc[dt][i] *= f;
        }

        // ---- P A-frags from LDS (same-wave DS ops are in-order) --------
        v16bf pf[2];
#pragma unroll
        for (int c = 0; c < 2; ++c) {
            BfFrag u;
            u.h[0] = *(const v8bf*)&Ps[wave][l16][c * 32 + 8 * half];
            u.h[1] = *(const v8bf*)&Ps[wave][l16][c * 32 + 16 + 8 * half];
            pf[c] = u.v;
        }
        // ---- ctx += P * V ---------------------------------------------
#pragma unroll
        for (int dt = 0; dt < 4; ++dt) {
            int d = dt * 16 + l16;
#pragma unroll
            for (int c = 0; c < 2; ++c) {          // two 32-wide key chunks
                BfFrag u;
                u.h[0] = *(const v8bf*)&Vs[d][c * 32 + 8 * half];
                u.h[1] = *(const v8bf*)&Vs[d][c * 32 + 16 + 8 * half];
                accc[dt] = wmma_bf16(pf[c], u.v, accc[dt]);
            }
        }
        __syncthreads();
    }

    // ---- finalize: ctx f32 into (B,S,D_MODEL) --------------------------
#pragma unroll
    for (int dt = 0; dt < 4; ++dt)
#pragma unroll
        for (int i = 0; i < 8; ++i) {
            int s = q0 + wq + 8 * half + i;
            int col = h * D_K + dt * 16 + l16;
            ctx[((size_t)(b * SEQ + s)) * D_MODEL + col] = accc[dt][i] / li[i];
        }
}

// ---------------------------------------------------------------------------
// Row LayerNorm over D_MODEL=1024; one block per row.
// ---------------------------------------------------------------------------
__global__ __launch_bounds__(256)
void layernorm_kernel(const float* __restrict__ res, const float* __restrict__ gamma,
                      const float* __restrict__ beta, float* __restrict__ out)
{
    __shared__ float smem[16];
    const int row = blockIdx.x;
    const float* r = res + (size_t)row * D_MODEL;
    float x[4], s = 0.f, s2 = 0.f;
#pragma unroll
    for (int j = 0; j < 4; ++j) {
        x[j] = r[threadIdx.x + j * 256];
        s += x[j]; s2 += x[j] * x[j];
    }
#pragma unroll
    for (int msk = 16; msk >= 1; msk >>= 1) {
        s  += __shfl_xor(s,  msk, 32);
        s2 += __shfl_xor(s2, msk, 32);
    }
    int wave = threadIdx.x >> 5, lane = threadIdx.x & 31;
    if (lane == 0) { smem[wave] = s; smem[8 + wave] = s2; }
    __syncthreads();
    if (wave == 0) {
        float a  = (lane < 8) ? smem[lane] : 0.f;
        float b2 = (lane < 8) ? smem[8 + lane] : 0.f;
#pragma unroll
        for (int msk = 4; msk >= 1; msk >>= 1) {
            a  += __shfl_xor(a,  msk, 32);
            b2 += __shfl_xor(b2, msk, 32);
        }
        if (lane == 0) { smem[0] = a; smem[1] = b2; }
    }
    __syncthreads();
    float mean = smem[0] * (1.f / D_MODEL);
    float var  = smem[1] * (1.f / D_MODEL) - mean * mean;
    float inv  = rsqrtf(var + 1e-5f);
#pragma unroll
    for (int j = 0; j < 4; ++j) {
        int c = threadIdx.x + j * 256;
        out[(size_t)row * D_MODEL + c] = (x[j] - mean) * inv * gamma[c] + beta[c];
    }
}

// ---------------------------------------------------------------------------
// Host entry
// ---------------------------------------------------------------------------
extern "C" void kernel_launch(void* const* d_in, const int* in_sizes, int n_in,
                              void* d_out, int out_size, void* d_ws, size_t ws_size,
                              hipStream_t stream)
{
    (void)in_sizes; (void)n_in; (void)out_size; (void)ws_size;
    const float* x     = (const float*)d_in[0];
    const float* Wqkv  = (const float*)d_in[1];
    const float* bqkv  = (const float*)d_in[2];
    const float* Wout  = (const float*)d_in[3];
    const float* bout  = (const float*)d_in[4];
    const float* gamma = (const float*)d_in[5];
    const float* beta  = (const float*)d_in[6];
    float* out = (float*)d_out;

    // workspace layout
    const size_t QKV_ELEMS = (size_t)BATCH * HEADS * SEQ * D_K;   // 8,388,608
    char* ws = (char*)d_ws;
    __bf16* qb  = (__bf16*)ws;                       ws += QKV_ELEMS * 2;
    __bf16* kb  = (__bf16*)ws;                       ws += QKV_ELEMS * 2;
    __bf16* vb  = (__bf16*)ws;                       ws += QKV_ELEMS * 2;
    float*  ctx = (float*)ws;                        ws += (size_t)ROWS * D_MODEL * 4;
    float*  res = (float*)ws;

    dim3 blk(256);

    // 1) fused QKV projection: (8192x1024) x (1024x3072)
    gemm_bf16_wmma<0><<<dim3(3 * D_MODEL / 128, ROWS / 128), blk, 0, stream>>>(
        x, Wqkv, bqkv, nullptr, qb, kb, vb, nullptr, ROWS, 3 * D_MODEL, D_MODEL);

    // 2) flash attention
    flash_attn_wmma<<<dim3(BATCH * HEADS * (SEQ / 128)), blk, 0, stream>>>(
        qb, kb, vb, ctx);

    // 3) out projection + bias + residual
    gemm_bf16_wmma<1><<<dim3(D_MODEL / 128, ROWS / 128), blk, 0, stream>>>(
        ctx, Wout, bout, x, nullptr, nullptr, nullptr, res, ROWS, D_MODEL, D_MODEL);

    // 4) layernorm
    layernorm_kernel<<<dim3(ROWS), blk, 0, stream>>>(res, gamma, beta, out);
}
